// RNN_75565654606177
// MI455X (gfx1250) — compile-verified, matrix operand order checked by reading, not run on hardware
//
#include <hip/hip_runtime.h>
#include <hip/hip_bf16.h>
#include <math.h>

#define IN_DIM 256
#define HID    512
#define OUT_DIM 256
#define BATCH  64
#define T_STEPS 2048
#define MROWS  (BATCH * T_STEPS)   // 131072

typedef unsigned short u16;
typedef __attribute__((ext_vector_type(16))) __bf16 bf16x16;
typedef __attribute__((ext_vector_type(8)))  __bf16 bf16x8;
typedef __attribute__((ext_vector_type(8)))  float  v8f;
typedef __attribute__((ext_vector_type(4)))  unsigned int uintx4;

// fp32 -> bf16, round-to-nearest-even
static __device__ __forceinline__ u16 f2bf(float f) {
    unsigned u = __builtin_bit_cast(unsigned, f);
    unsigned r = u + 0x7FFFu + ((u >> 16) & 1u);
    return (u16)(r >> 16);
}

// Assemble a 16-element bf16 WMMA fragment from two 16-byte chunks.
static __device__ __forceinline__ bf16x16 make_frag(uintx4 lo, uintx4 hi) {
    bf16x8 a = __builtin_bit_cast(bf16x8, lo);
    bf16x8 b = __builtin_bit_cast(bf16x8, hi);
    bf16x16 r;
#pragma unroll
    for (int i = 0; i < 8; ++i) { r[i] = a[i]; r[8 + i] = b[i]; }
    return r;
}

static __device__ __forceinline__ uintx4 ld16(const u16* p) {
    return *(const uintx4*)p;
}

// ---------------------------------------------------------------- prep kernels
__global__ void convert_bf16_kernel(const float* __restrict__ src,
                                    u16* __restrict__ dst, int n) {
    int i = blockIdx.x * blockDim.x + threadIdx.x;
    if (i < n) dst[i] = f2bf(src[i]);
}

__global__ void bias_sum_kernel(const float* __restrict__ a,
                                const float* __restrict__ b,
                                float* __restrict__ out, int n) {
    int i = blockIdx.x * blockDim.x + threadIdx.x;
    if (i < n) out[i] = a[i] + b[i];
}

// ---------------------------------------------------------------- kernel 2: xproj
// xp[m][h] = sum_i x[m][i] * Whx[h][i] + (Whx_b[h] + Whh_b[h])
// One WG = one M-tile of 16 rows staged in LDS; 8 waves x 4 N-tiles cover HID=512.
__global__ __launch_bounds__(256) void xproj_kernel(
    const float* __restrict__ x,     // (MROWS, IN_DIM) fp32
    const u16*  __restrict__ whx,    // (HID, IN_DIM) bf16
    const float* __restrict__ bsum,  // (HID)
    float* __restrict__ xp)          // (MROWS, HID) fp32
{
    __shared__ u16 sA[16][IN_DIM + 8];   // +8 u16 pad -> conflict-free b128 reads
    const int mbase = blockIdx.x * 16;
    const int tid = threadIdx.x;

    // stage + convert the 16x256 x tile
    for (int i = tid; i < 16 * IN_DIM; i += 256) {
        int r = i >> 8, c = i & (IN_DIM - 1);
        sA[r][c] = f2bf(x[(size_t)(mbase + r) * IN_DIM + c]);
    }
    __syncthreads();

    const int wave = tid >> 5, lane = tid & 31;
    const int mrow = lane & 15;
    const int ksel = (lane >> 4) << 3;           // 0 or 8 (A/B K-half select)
    const int rsel = (lane >> 4) << 3;           // 0 or 8 (C/D row-half select)
    const u16* sArow = &sA[mrow][0];

    for (int j = 0; j < 4; ++j) {
        const int n0 = (wave * 4 + j) * 16;
        const int ncol = n0 + (lane & 15);
        const float bias = bsum[ncol];
        v8f acc;
#pragma unroll
        for (int v = 0; v < 8; ++v) acc[v] = bias;

        const u16* brow = whx + (size_t)ncol * IN_DIM;
#pragma unroll
        for (int k0 = 0; k0 < IN_DIM; k0 += 32) {
            uintx4 a0 = ld16(sArow + k0 + ksel);
            uintx4 a1 = ld16(sArow + k0 + 16 + ksel);
            uintx4 b0 = ld16(brow + k0 + ksel);
            uintx4 b1 = ld16(brow + k0 + 16 + ksel);
            acc = __builtin_amdgcn_wmma_f32_16x16x32_bf16(
                false, make_frag(a0, a1), false, make_frag(b0, b1),
                (short)0, acc, false, false);
        }
#pragma unroll
        for (int v = 0; v < 8; ++v)
            xp[(size_t)(mbase + rsel + v) * HID + ncol] = acc[v];
    }
}

// ---------------------------------------------------------------- kernel 3: recurrence
// One persistent 512-thread WG per batch group of 16 (grid = 4).
// h (16x512 bf16) lives in LDS. 16 waves, each owning TWO 16-column tiles:
//   - Whh fragments for K in [0,256) are preloaded into VGPRs (64 VGPRs/lane
//     per tile) -> zero per-step traffic for half the weight matrix.
//   - K in [256,512) streams from the L2-resident bf16 Whh copy.
//   - A fragments (h) are loaded once per k-step and reused for both tiles,
//     halving LDS read amplification.
__global__ __launch_bounds__(512) void rnn_step_kernel(
    const float* __restrict__ h0,    // (BATCH, HID) fp32
    const float* __restrict__ xp,    // (BATCH, T, HID) fp32 (biases folded in)
    const u16*  __restrict__ whh,    // (HID, HID) bf16
    u16*  __restrict__ hs,           // (BATCH, T, HID) bf16 out
    float* __restrict__ hT)          // (BATCH, HID) fp32 out
{
    __shared__ u16 sH[16][HID + 8];
    const int bg = blockIdx.x;               // batch group 0..3
    const int tid = threadIdx.x;

    // init h state from h_0
    for (int i = tid; i < 16 * HID; i += 512) {
        int r = i >> 9, c = i & (HID - 1);
        sH[r][c] = f2bf(h0[(size_t)(bg * 16 + r) * HID + c]);
    }

    const int wave = tid >> 5, lane = tid & 31;
    const int ln   = lane & 15;
    const int ksel = (lane >> 4) << 3;       // A/B K-half select
    const int rsel = (lane >> 4) << 3;       // C/D row-half select
    const int ncA  = wave * 16 + ln;         // tile A column (0..255)
    const int ncB  = 256 + wave * 16 + ln;   // tile B column (256..511)
    const u16* wrowA = whh + (size_t)ncA * HID;
    const u16* wrowB = whh + (size_t)ncB * HID;
    const u16* hrow  = &sH[ln][0];

    // Preload Whh fragments for K in [0,256) into registers (once).
    bf16x16 bfA[8], bfB[8];
#pragma unroll
    for (int kk = 0; kk < 8; ++kk) {
        bfA[kk] = make_frag(ld16(wrowA + kk * 32 + ksel),
                            ld16(wrowA + kk * 32 + 16 + ksel));
        bfB[kk] = make_frag(ld16(wrowB + kk * 32 + ksel),
                            ld16(wrowB + kk * 32 + 16 + ksel));
    }
    __syncthreads();

    for (int t = 0; t < T_STEPS; ++t) {
        // accumulators start at xproj_t (includes both biases)
        v8f accA, accB;
#pragma unroll
        for (int v = 0; v < 8; ++v) {
            size_t row = (size_t)(bg * 16 + rsel + v) * T_STEPS + t;
            accA[v] = xp[row * HID + ncA];
            accB[v] = xp[row * HID + ncB];
        }
        // warm L2/WGP$ for next step's strided xp lines (global_prefetch_b8)
        if (t + 1 < T_STEPS) {
#pragma unroll
            for (int v = 0; v < 8; ++v) {
                size_t row = (size_t)(bg * 16 + rsel + v) * T_STEPS + t + 1;
                __builtin_prefetch(&xp[row * HID + ncA], 0, 0);
                __builtin_prefetch(&xp[row * HID + ncB], 0, 0);
            }
        }

        // K in [0,256): weights from VGPRs, A from LDS (shared by both tiles)
#pragma unroll
        for (int kk = 0; kk < 8; ++kk) {
            bf16x16 af = make_frag(ld16(hrow + kk * 32 + ksel),
                                   ld16(hrow + kk * 32 + 16 + ksel));
            accA = __builtin_amdgcn_wmma_f32_16x16x32_bf16(
                false, af, false, bfA[kk], (short)0, accA, false, false);
            accB = __builtin_amdgcn_wmma_f32_16x16x32_bf16(
                false, af, false, bfB[kk], (short)0, accB, false, false);
        }
        // K in [256,512): weights stream from L2
#pragma unroll 2
        for (int k0 = 256; k0 < 512; k0 += 32) {
            bf16x16 af = make_frag(ld16(hrow + k0 + ksel),
                                   ld16(hrow + k0 + 16 + ksel));
            bf16x16 bA = make_frag(ld16(wrowA + k0 + ksel),
                                   ld16(wrowA + k0 + 16 + ksel));
            bf16x16 bB = make_frag(ld16(wrowB + k0 + ksel),
                                   ld16(wrowB + k0 + 16 + ksel));
            accA = __builtin_amdgcn_wmma_f32_16x16x32_bf16(
                false, af, false, bA, (short)0, accA, false, false);
            accB = __builtin_amdgcn_wmma_f32_16x16x32_bf16(
                false, af, false, bB, (short)0, accB, false, false);
        }

        float hvA[8], hvB[8];
        u16 hbA[8], hbB[8];
#pragma unroll
        for (int v = 0; v < 8; ++v) {
            hvA[v] = tanhf(accA[v]); hbA[v] = f2bf(hvA[v]);
            hvB[v] = tanhf(accB[v]); hbB[v] = f2bf(hvB[v]);
        }

        __syncthreads();   // all waves finished reading sH for this step
#pragma unroll
        for (int v = 0; v < 8; ++v) {
            sH[rsel + v][ncA] = hbA[v];
            sH[rsel + v][ncB] = hbB[v];
            size_t row = (size_t)(bg * 16 + rsel + v) * T_STEPS + t;
            hs[row * HID + ncA] = hbA[v];
            hs[row * HID + ncB] = hbB[v];
        }
        if (t == T_STEPS - 1) {
#pragma unroll
            for (int v = 0; v < 8; ++v) {
                hT[(size_t)(bg * 16 + rsel + v) * HID + ncA] = hvA[v];
                hT[(size_t)(bg * 16 + rsel + v) * HID + ncB] = hvB[v];
            }
        }
        __syncthreads();   // sH fully updated before next step's reads
    }
}

// ---------------------------------------------------------------- kernel 4: y GEMM
// y[m][o] = sum_h hs[m][h] * Wyh[o][h] + Wyh_b[o]
__global__ __launch_bounds__(256) void ygemm_kernel(
    const u16*  __restrict__ hs,    // (MROWS, HID) bf16
    const u16*  __restrict__ wyh,   // (OUT_DIM, HID) bf16
    const float* __restrict__ yb,   // (OUT_DIM) fp32
    float* __restrict__ y)          // (MROWS, OUT_DIM) fp32
{
    __shared__ u16 sA[16][HID + 8];
    const int mbase = blockIdx.x * 16;
    const int tid = threadIdx.x;

    // stage 16x512 bf16 hs tile (rows are contiguous -> flat b128 copy)
    const uintx4* src = (const uintx4*)(hs + (size_t)mbase * HID);
    for (int i = tid; i < 16 * HID / 8; i += 256) {
        int r = i >> 6, c = i & 63;
        *(uintx4*)(&sA[r][c * 8]) = src[i];
    }
    __syncthreads();

    const int wave = tid >> 5, lane = tid & 31;
    const int ksel = (lane >> 4) << 3;
    const int rsel = (lane >> 4) << 3;
    const u16* sArow = &sA[lane & 15][0];

    for (int j = 0; j < 2; ++j) {
        const int n0 = (wave * 2 + j) * 16;
        const int ncol = n0 + (lane & 15);
        const float bias = yb[ncol];
        v8f acc;
#pragma unroll
        for (int v = 0; v < 8; ++v) acc[v] = bias;

        const u16* brow = wyh + (size_t)ncol * HID;
#pragma unroll 4
        for (int k0 = 0; k0 < HID; k0 += 32) {
            uintx4 a0 = ld16(sArow + k0 + ksel);
            uintx4 a1 = ld16(sArow + k0 + 16 + ksel);
            uintx4 b0 = ld16(brow + k0 + ksel);
            uintx4 b1 = ld16(brow + k0 + 16 + ksel);
            acc = __builtin_amdgcn_wmma_f32_16x16x32_bf16(
                false, make_frag(a0, a1), false, make_frag(b0, b1),
                (short)0, acc, false, false);
        }
#pragma unroll
        for (int v = 0; v < 8; ++v)
            y[(size_t)(mbase + rsel + v) * OUT_DIM + ncol] = acc[v];
    }
}

// ---------------------------------------------------------------- launch
extern "C" void kernel_launch(void* const* d_in, const int* in_sizes, int n_in,
                              void* d_out, int out_size, void* d_ws, size_t ws_size,
                              hipStream_t stream) {
    (void)in_sizes; (void)n_in; (void)out_size; (void)ws_size;

    const float* x     = (const float*)d_in[0];   // (B,T,IN)
    const float* h0    = (const float*)d_in[1];   // (B,HID)
    const float* Whh_w = (const float*)d_in[2];   // (HID,HID)
    const float* Whh_b = (const float*)d_in[3];   // (HID)
    const float* Whx_w = (const float*)d_in[4];   // (HID,IN)
    const float* Whx_b = (const float*)d_in[5];   // (HID)
    const float* Wyh_w = (const float*)d_in[6];   // (OUT,HID)
    const float* Wyh_b = (const float*)d_in[7];   // (OUT)

    float* y  = (float*)d_out;                         // (B,T,OUT)
    float* hT = y + (size_t)MROWS * OUT_DIM;           // (B,HID)

    // workspace layout
    char* ws = (char*)d_ws;
    float* xp   = (float*)ws;                                      // 256 MB
    size_t off  = (size_t)MROWS * HID * sizeof(float);
    u16* hs     = (u16*)(ws + off);                                // 128 MB
    off        += (size_t)MROWS * HID * sizeof(u16);
    u16* whh16  = (u16*)(ws + off); off += (size_t)HID * HID * sizeof(u16);
    u16* whx16  = (u16*)(ws + off); off += (size_t)HID * IN_DIM * sizeof(u16);
    u16* wyh16  = (u16*)(ws + off); off += (size_t)OUT_DIM * HID * sizeof(u16);
    float* bsum = (float*)(ws + off);

    // 1) weight conversions + fused bias
    convert_bf16_kernel<<<(HID * HID + 255) / 256, 256, 0, stream>>>(Whh_w, whh16, HID * HID);
    convert_bf16_kernel<<<(HID * IN_DIM + 255) / 256, 256, 0, stream>>>(Whx_w, whx16, HID * IN_DIM);
    convert_bf16_kernel<<<(OUT_DIM * HID + 255) / 256, 256, 0, stream>>>(Wyh_w, wyh16, OUT_DIM * HID);
    bias_sum_kernel<<<(HID + 255) / 256, 256, 0, stream>>>(Whx_b, Whh_b, bsum, HID);

    // 2) input projection GEMM (memory-bound, full-GPU)
    xproj_kernel<<<MROWS / 16, 256, 0, stream>>>(x, whx16, bsum, xp);

    // 3) sequential recurrence: 4 persistent WGs, one per batch group of 16
    rnn_step_kernel<<<4, 512, 0, stream>>>(h0, xp, whh16, hs, hT);

    // 4) output GEMM (memory-bound, full-GPU)
    ygemm_kernel<<<MROWS / 16, 256, 0, stream>>>(hs, wyh16, Wyh_b, y);
}
